// TransformerEncoderLayer_5729486373304
// MI455X (gfx1250) — matrix operand order repaired and checked by reading, not compiled, not run on hardware
//
#include <hip/hip_runtime.h>

// ---------------------------------------------------------------------------
// Transformer encoder layer for MI455X (gfx1250), wave32 + WMMA f16 +
// async global->LDS double-buffered GEMM pipeline.
// B=8, S=1024, D=1024, H=16, DK=64, F=4096.
// ---------------------------------------------------------------------------

typedef _Float16 half_t;
typedef __attribute__((ext_vector_type(8)))  _Float16 v8h;
typedef __attribute__((ext_vector_type(16))) _Float16 v16h;
typedef __attribute__((ext_vector_type(8)))  float    v8f;

// ---- WMMA helpers ---------------------------------------------------------

__device__ __forceinline__ v8f wmma_f16(v16h a, v16h b, v8f c) {
  // (neg_a, A, neg_b, B, c_mod, C, reuse_a, reuse_b)
  return __builtin_amdgcn_wmma_f32_16x16x32_f16(false, a, false, b, (short)0, c,
                                                false, false);
}

// A-matrix 16x32 f16 fragment from row-major storage (ld in halves).
// ISA layout: lanes 0-15 -> M=lane, K groups {0-7,16-23};
//             lanes 16-31 -> M=lane-16, K groups {8-15,24-31}.
__device__ __forceinline__ v16h frag_a(const half_t* p, int ld) {
  const int lane = threadIdx.x & 31;
  const int m = lane & 15, hi = lane >> 4;
  const half_t* q = p + (size_t)m * ld + hi * 8;
  v8h g0 = *(const v8h*)q;
  v8h g1 = *(const v8h*)(q + 16);
  return __builtin_shufflevector(g0, g1, 0, 1, 2, 3, 4, 5, 6, 7,
                                 8, 9, 10, 11, 12, 13, 14, 15);
}

// B-matrix 32x16 f16 fragment from N-major (transposed) storage:
// element (k,n) stored at p[n*ld + k]. ISA layout: fixed N per lane
// (n = lane&15), lanes 0-15 hold K groups {0-7,16-23}, lanes 16-31 hold {8-15,24-31}.
__device__ __forceinline__ v16h frag_b(const half_t* p, int ld) {
  const int lane = threadIdx.x & 31;
  const int n = lane & 15, hi = lane >> 4;
  const half_t* q = p + (size_t)n * ld + hi * 8;
  v8h g0 = *(const v8h*)q;
  v8h g1 = *(const v8h*)(q + 16);
  return __builtin_shufflevector(g0, g1, 0, 1, 2, 3, 4, 5, 6, 7,
                                 8, 9, 10, 11, 12, 13, 14, 15);
}

// Async global->LDS copy of 16 bytes (8 halves), tracked by ASYNCcnt.
// LDS byte address = low 32 bits of the generic shared pointer (FLAT
// aperture truncation, ISA 10.2).
__device__ __forceinline__ void async_copy16(const half_t* g, half_t* l) {
  unsigned lds_addr = (unsigned)(uintptr_t)l;
  asm volatile("global_load_async_to_lds_b128 %0, %1, off"
               :: "v"(lds_addr), "v"(g)
               : "memory");
}

// ---- Elementwise conversion / transpose kernels ---------------------------

__global__ void cvt_f32_f16_kernel(const float* __restrict__ in,
                                   half_t* __restrict__ out, int n) {
  int i = blockIdx.x * blockDim.x + threadIdx.x;
  if (i < n) out[i] = (half_t)in[i];
}

// out[bt][c][r] = (f16) in[bt][r][c]   (i enumerates out flat, r fastest)
__global__ void transpose_cvt_kernel(const float* __restrict__ in,
                                     half_t* __restrict__ out,
                                     int R, int C, int total) {
  int i = blockIdx.x * blockDim.x + threadIdx.x;
  if (i >= total) return;
  int r = i % R;
  int rem = i / R;
  int c = rem % C;
  int bt = rem / C;
  out[i] = (half_t)in[((size_t)bt * R + r) * C + c];
}

// vt[b][h][dk][s] = v[b*S+s][h*64+dk]   (f16 -> f16)
__global__ void transpose_v_kernel(const half_t* __restrict__ v,
                                   half_t* __restrict__ vt) {
  int i = blockIdx.x * blockDim.x + threadIdx.x;  // exactly B*H*64*S threads
  int s  = i & 1023;
  int dk = (i >> 10) & 63;
  int h  = (i >> 16) & 15;
  int b  = i >> 20;
  vt[i] = v[(size_t)(b * 1024 + s) * 1024 + h * 64 + dk];
}

// ---- Tiled WMMA GEMM:  C[M,N] = A[M,K] * Bt[N,K]^T + bias, opt ReLU -------
// Double-buffered LDS fed by GLOBAL_LOAD_ASYNC_TO_LDS_B128 (ASYNCcnt),
// overlapping global fetch of tile i+1 with WMMA on tile i.

template <bool RELU, typename OUT>
__global__ __launch_bounds__(256) void gemm_wmma_kernel(
    const half_t* __restrict__ A, const half_t* __restrict__ Bt,
    const float* __restrict__ bias, OUT* __restrict__ C,
    int M, int N, int K) {
  __shared__ alignas(16) half_t As[2][128][40];
  __shared__ alignas(16) half_t Bs[2][128][40];

  const int tn0 = blockIdx.x * 128;
  const int tm0 = blockIdx.y * 128;
  const int t = threadIdx.x;
  const int w = t >> 5;
  const int wm = (w >> 1) * 32;  // 4 wave-rows of 32
  const int wn = (w & 1) * 64;   // 2 wave-cols of 64

  v8f acc[2][4];
#pragma unroll
  for (int i = 0; i < 2; i++)
#pragma unroll
    for (int j = 0; j < 4; j++)
#pragma unroll
      for (int e = 0; e < 8; e++) acc[i][j][e] = 0.0f;

  const int lrow = t >> 1;
  const int lcol = (t & 1) * 16;

  // issue the 4 async 16B copies (A-tile + B-tile) for k-step kb into buf
  auto issue = [&](int buf, int kb) {
    const half_t* ag = A + (size_t)(tm0 + lrow) * K + kb + lcol;
    async_copy16(ag, &As[buf][lrow][lcol]);
    async_copy16(ag + 8, &As[buf][lrow][lcol + 8]);
    const half_t* bg = Bt + (size_t)(tn0 + lrow) * K + kb + lcol;
    async_copy16(bg, &Bs[buf][lrow][lcol]);
    async_copy16(bg + 8, &Bs[buf][lrow][lcol + 8]);
  };

  const int steps = K >> 5;
  issue(0, 0);
  for (int i = 0; i < steps; i++) {
    const int buf = i & 1;
    if (i + 1 < steps) {
      issue(buf ^ 1, (i + 1) << 5);
      // async loads retire in order: <=4 outstanding => tile i has landed
      asm volatile("s_wait_asynccnt 0x4" ::: "memory");
    } else {
      asm volatile("s_wait_asynccnt 0x0" ::: "memory");
    }
    __syncthreads();

    v16h a0 = frag_a(&As[buf][wm][0], 40);
    v16h a1 = frag_a(&As[buf][wm + 16][0], 40);
#pragma unroll
    for (int j = 0; j < 4; j++) {
      v16h b = frag_b(&Bs[buf][wn + j * 16][0], 40);
      acc[0][j] = wmma_f16(a0, b, acc[0][j]);
      acc[1][j] = wmma_f16(a1, b, acc[1][j]);
    }
    __syncthreads();  // all waves done reading buf before it is refilled
  }

  // C/D layout: lane n = lane&15 holds column n; vgpr r -> row (lane>>4)*8+r
  const int lane = t & 31;
  const int n = lane & 15, hi = lane >> 4;
#pragma unroll
  for (int i = 0; i < 2; i++)
#pragma unroll
    for (int j = 0; j < 4; j++) {
      const int c0 = tn0 + wn + j * 16 + n;
      const float bv = bias[c0];
#pragma unroll
      for (int r = 0; r < 8; r++) {
        const int row = tm0 + wm + i * 16 + hi * 8 + r;
        float v = acc[i][j][r] + bv;
        if (RELU) v = fmaxf(v, 0.0f);
        C[(size_t)row * N + c0] = static_cast<OUT>(v);
      }
    }
}

// ---- Flash attention: one wave per 16 query rows per (b,h) ----------------
// Q,K: [B*S, H*64] f16 row-major.  Vt: [B,H,64,S] f16.  O: [B*S, H*64] f16.

__global__ __launch_bounds__(256) void attention_kernel(
    const half_t* __restrict__ Q, const half_t* __restrict__ Kh,
    const half_t* __restrict__ Vt, half_t* __restrict__ O) {
  __shared__ alignas(16) half_t P[8][16][40];  // per-wave P-tile scratch

  const int b = blockIdx.z, h = blockIdx.y;
  const int w = threadIdx.x >> 5;
  const int m0 = (blockIdx.x * 8 + w) * 16;  // query row block
  const int lane = threadIdx.x & 31;
  const int nn = lane & 15, hi = lane >> 4;

  const half_t* qp = Q + (size_t)(b * 1024 + m0) * 1024 + h * 64;
  v16h q0 = frag_a(qp, 1024);       // dk 0..31
  v16h q1 = frag_a(qp + 32, 1024);  // dk 32..63

  v8f oacc[4];
#pragma unroll
  for (int d = 0; d < 4; d++)
#pragma unroll
    for (int e = 0; e < 8; e++) oacc[d][e] = 0.0f;
  float mrow[8], lrow[8];
#pragma unroll
  for (int r = 0; r < 8; r++) { mrow[r] = -1e30f; lrow[r] = 0.0f; }

  for (int j = 0; j < 32; j++) {  // 32 key tiles of 32 keys
    const half_t* kp = Kh + (size_t)(b * 1024 + j * 32) * 1024 + h * 64;
    v8f s0, s1;
#pragma unroll
    for (int e = 0; e < 8; e++) { s0[e] = 0.0f; s1[e] = 0.0f; }
    s0 = wmma_f16(q0, frag_b(kp, 1024), s0);                    // keys 0-15
    s0 = wmma_f16(q1, frag_b(kp + 32, 1024), s0);
    s1 = wmma_f16(q0, frag_b(kp + 16 * 1024, 1024), s1);        // keys 16-31
    s1 = wmma_f16(q1, frag_b(kp + 16 * 1024 + 32, 1024), s1);

#pragma unroll
    for (int r = 0; r < 8; r++) {
      float sa = s0[r] * 0.125f, sb = s1[r] * 0.125f;  // 1/sqrt(64)
      float mx = fmaxf(sa, sb);
#pragma unroll
      for (int msk = 1; msk < 16; msk <<= 1)
        mx = fmaxf(mx, __shfl_xor(mx, msk, 32));
      const float mnew = fmaxf(mrow[r], mx);
      const float alpha = __expf(mrow[r] - mnew);
      const float pa = __expf(sa - mnew);
      const float pb = __expf(sb - mnew);
      float rs = pa + pb;
#pragma unroll
      for (int msk = 1; msk < 16; msk <<= 1) rs += __shfl_xor(rs, msk, 32);
      lrow[r] = lrow[r] * alpha + rs;
      mrow[r] = mnew;
#pragma unroll
      for (int d = 0; d < 4; d++) oacc[d][r] *= alpha;
      const int m = hi * 8 + r;
      P[w][m][nn] = (half_t)pa;
      P[w][m][16 + nn] = (half_t)pb;
    }
    // order LDS stores (C layout) before A-layout reload (same-wave, in-order)
    asm volatile("s_wait_dscnt 0" ::: "memory");
    v16h pfrag = frag_a(&P[w][0][0], 40);  // 16x32, K = key-in-tile
    const half_t* vp = Vt + (size_t)(b * 16 + h) * 64 * 1024 + j * 32;
#pragma unroll
    for (int d = 0; d < 4; d++) {
      v16h vf = frag_b(vp + (size_t)d * 16 * 1024, 1024);  // N=dk, K=key
      oacc[d] = wmma_f16(pfrag, vf, oacc[d]);
    }
  }

  half_t* op = O + (size_t)(b * 1024 + m0) * 1024 + h * 64;
#pragma unroll
  for (int r = 0; r < 8; r++) {
    const float inv = 1.0f / lrow[r];
    const int m = hi * 8 + r;
#pragma unroll
    for (int d = 0; d < 4; d++)
      op[(size_t)m * 1024 + d * 16 + nn] = (half_t)(oacc[d][r] * inv);
  }
}

// ---- Residual add + LayerNorm over D=1024 ---------------------------------

__global__ __launch_bounds__(256) void add_ln_kernel(
    const float* __restrict__ a, const float* __restrict__ bsrc,
    const float* __restrict__ g, const float* __restrict__ be,
    float* __restrict__ out, half_t* __restrict__ outh) {
  const int row = blockIdx.x;
  const int t = threadIdx.x;
  __shared__ float red1[256], red2[256];
  float vals[4];
  float s1 = 0.0f, s2 = 0.0f;
#pragma unroll
  for (int i = 0; i < 4; i++) {
    const int c = t + i * 256;
    const float v = a[(size_t)row * 1024 + c] + bsrc[(size_t)row * 1024 + c];
    vals[i] = v;
    s1 += v;
    s2 += v * v;
  }
  red1[t] = s1;
  red2[t] = s2;
  __syncthreads();
  for (int off = 128; off > 0; off >>= 1) {
    if (t < off) {
      red1[t] += red1[t + off];
      red2[t] += red2[t + off];
    }
    __syncthreads();
  }
  const float mean = red1[0] * (1.0f / 1024.0f);
  const float var = red2[0] * (1.0f / 1024.0f) - mean * mean;
  const float rstd = rsqrtf(var + 1e-5f);
#pragma unroll
  for (int i = 0; i < 4; i++) {
    const int c = t + i * 256;
    const float y = (vals[i] - mean) * rstd * g[c] + be[c];
    out[(size_t)row * 1024 + c] = y;
    if (outh) outh[(size_t)row * 1024 + c] = (half_t)y;
  }
}

// ---- Host-side orchestration ----------------------------------------------

static constexpr size_t MB = 1024ull * 1024ull;
static constexpr size_t OFF_SRCH = 0;         // 8192x1024 f16 (16MB)
static constexpr size_t OFF_WQ = 16 * MB;     // [1024][1024] f16 N-major (2MB)
static constexpr size_t OFF_WK = 18 * MB;
static constexpr size_t OFF_WV = 20 * MB;
static constexpr size_t OFF_WO = 22 * MB;
static constexpr size_t OFF_W1 = 24 * MB;     // [4096][1024] f16 (8MB)
static constexpr size_t OFF_W2 = 32 * MB;     // [1024][4096] f16 (8MB)
static constexpr size_t OFF_Q  = 40 * MB;     // 8192x1024 f16 (16MB)
static constexpr size_t OFF_K  = 56 * MB;
static constexpr size_t OFF_V  = 72 * MB;
static constexpr size_t OFF_VT = 88 * MB;     // [B,H,64,S] f16 (16MB)
static constexpr size_t OFF_O  = 104 * MB;    // attn concat f16 (16MB)
static constexpr size_t OFF_AO = 120 * MB;    // attn proj f32 (32MB)
static constexpr size_t OFF_X  = 152 * MB;    // post-LN1 f32 (32MB)
static constexpr size_t OFF_XH = 184 * MB;    // post-LN1 f16 (16MB)
static constexpr size_t OFF_FF = 200 * MB;    // relu(fc1) f16 8192x4096 (64MB)
static constexpr size_t OFF_F2 = 264 * MB;    // fc2 f32 (32MB)  -> end 296MB

extern "C" void kernel_launch(void* const* d_in, const int* in_sizes, int n_in,
                              void* d_out, int out_size, void* d_ws,
                              size_t ws_size, hipStream_t stream) {
  const float* src = (const float*)d_in[0];
  const float* Wq = (const float*)d_in[1];
  const float* bq = (const float*)d_in[2];
  const float* Wk = (const float*)d_in[3];
  const float* bk = (const float*)d_in[4];
  const float* Wv = (const float*)d_in[5];
  const float* bv = (const float*)d_in[6];
  const float* Wo = (const float*)d_in[7];
  const float* bo = (const float*)d_in[8];
  const float* g1 = (const float*)d_in[9];
  const float* be1 = (const float*)d_in[10];
  const float* W1 = (const float*)d_in[11];
  const float* b1 = (const float*)d_in[12];
  const float* W2 = (const float*)d_in[13];
  const float* b2 = (const float*)d_in[14];
  const float* g2 = (const float*)d_in[15];
  const float* be2 = (const float*)d_in[16];

  char* ws = (char*)d_ws;
  half_t* srch = (half_t*)(ws + OFF_SRCH);
  half_t* Wqh = (half_t*)(ws + OFF_WQ);
  half_t* Wkh = (half_t*)(ws + OFF_WK);
  half_t* Wvh = (half_t*)(ws + OFF_WV);
  half_t* Woh = (half_t*)(ws + OFF_WO);
  half_t* W1h = (half_t*)(ws + OFF_W1);
  half_t* W2h = (half_t*)(ws + OFF_W2);
  half_t* qh = (half_t*)(ws + OFF_Q);
  half_t* kh = (half_t*)(ws + OFF_K);
  half_t* vh = (half_t*)(ws + OFF_V);
  half_t* vth = (half_t*)(ws + OFF_VT);
  half_t* oh = (half_t*)(ws + OFF_O);
  float* aof = (float*)(ws + OFF_AO);
  float* xf = (float*)(ws + OFF_X);
  half_t* xh = (half_t*)(ws + OFF_XH);
  half_t* ffh = (half_t*)(ws + OFF_FF);
  float* f2f = (float*)(ws + OFF_F2);

  const dim3 blk(256);

  // 1) precision conversion + weight transposition (N-major for WMMA B)
  cvt_f32_f16_kernel<<<32768, blk, 0, stream>>>(src, srch, 8192 * 1024);
  transpose_cvt_kernel<<<4096, blk, 0, stream>>>(Wq, Wqh, 1024, 64, 1048576);
  transpose_cvt_kernel<<<4096, blk, 0, stream>>>(Wk, Wkh, 1024, 64, 1048576);
  transpose_cvt_kernel<<<4096, blk, 0, stream>>>(Wv, Wvh, 1024, 64, 1048576);
  transpose_cvt_kernel<<<4096, blk, 0, stream>>>(Wo, Woh, 1024, 1024, 1048576);
  transpose_cvt_kernel<<<16384, blk, 0, stream>>>(W1, W1h, 1024, 4096, 4194304);
  transpose_cvt_kernel<<<16384, blk, 0, stream>>>(W2, W2h, 4096, 1024, 4194304);

  // 2) Q/K/V projections  [8192,1024] x [1024,1024]
  gemm_wmma_kernel<false, half_t><<<dim3(8, 64), blk, 0, stream>>>(
      srch, Wqh, bq, qh, 8192, 1024, 1024);
  gemm_wmma_kernel<false, half_t><<<dim3(8, 64), blk, 0, stream>>>(
      srch, Wkh, bk, kh, 8192, 1024, 1024);
  gemm_wmma_kernel<false, half_t><<<dim3(8, 64), blk, 0, stream>>>(
      srch, Wvh, bv, vh, 8192, 1024, 1024);

  // 3) transpose V for PV B-fragments, then flash attention
  transpose_v_kernel<<<32768, blk, 0, stream>>>(vh, vth);
  attention_kernel<<<dim3(8, 16, 8), blk, 0, stream>>>(qh, kh, vth, oh);

  // 4) output projection + residual LN1
  gemm_wmma_kernel<false, float><<<dim3(8, 64), blk, 0, stream>>>(
      oh, Woh, bo, aof, 8192, 1024, 1024);
  add_ln_kernel<<<8192, blk, 0, stream>>>(src, aof, g1, be1, xf, xh);

  // 5) FFN + residual LN2 -> d_out (f32)
  gemm_wmma_kernel<true, half_t><<<dim3(32, 64), blk, 0, stream>>>(
      xh, W1h, b1, ffh, 8192, 4096, 1024);
  gemm_wmma_kernel<false, float><<<dim3(8, 64), blk, 0, stream>>>(
      ffh, W2h, b2, f2f, 8192, 1024, 4096);
  add_ln_kernel<<<8192, blk, 0, stream>>>(xf, f2f, g2, be2, (float*)d_out,
                                          (half_t*)nullptr);
}
